// OfficialUPFDModel_67800353734947
// MI455X (gfx1250) — matrix operand order repaired and checked by reading, not compiled, not run on hardware
//
#include <hip/hip_runtime.h>
#include <hip/hip_bf16.h>

typedef __attribute__((ext_vector_type(16))) _Float16 v16h;
typedef __attribute__((ext_vector_type(8)))  _Float16 v8h;
typedef __attribute__((ext_vector_type(8)))  float    v8f;

#define NUM_GRAPHS 2048
#define NODES_PER_GRAPH 256
#define N_NODES (NUM_GRAPHS * NODES_PER_GRAPH)   // 524288
#define N_EDGES (2 * N_NODES)                    // 1048576
#define F_IN 768
#define HID 128
#define KSTEPS (F_IN / 32)                       // 24
#define NTILES (HID / 16)                        // 8
#define NFRAGS (KSTEPS * NTILES)                 // 192

// ---------------------------------------------------------------------------
// Pre-swizzle a [F_IN, HID] f32 weight matrix into per-lane f16 B fragments.
// 16-bit B matrix 32x16 layout (wave32): lane l holds column n = l%16;
// half e of lane half kh holds K = kh*8 + e for e<8, K = 16 + kh*8 + (e-8)
// for e>=8 (matches the A fill). 16 halves per lane, stored contiguous
// so the GEMM loads each fragment with two b128 loads.
// ---------------------------------------------------------------------------
__global__ void swizzle_w_kernel(const float* __restrict__ W, _Float16* __restrict__ out) {
  int idx = blockIdx.x * 256 + threadIdx.x;       // 24*8*32*16 = 98304 elements
  if (idx >= NFRAGS * 32 * 16) return;
  int e = idx & 15;          // half index within lane fragment
  int l = (idx >> 4) & 31;   // lane
  int j = (idx >> 9) & 7;    // n-tile
  int t = idx >> 12;         // k-step
  int v  = e >> 1;
  int lo = e & 1;
  int k = t * 32 + (l >> 4) * 8 + (e >> 3) * 16 + (e & 7); (void)v; (void)lo;
  int n = j * 16 + (l & 15);
  out[idx] = (_Float16)W[k * HID + n];
}

// ---------------------------------------------------------------------------
// Zero a float buffer (grid-stride).
// ---------------------------------------------------------------------------
__global__ void zero_f32_kernel(float* __restrict__ p, long long n) {
  long long i = (long long)blockIdx.x * blockDim.x + threadIdx.x;
  long long stride = (long long)gridDim.x * blockDim.x;
  for (; i < n; i += stride) p[i] = 0.0f;
}

// ---------------------------------------------------------------------------
// WMMA GEMM, software-pipelined.
//   DUAL=true : Yl = X @ Bl, Yr = X @ Br (shared single pass over X rows)
//   RELU=true : Yl = relu(X @ Bl + bias)
// X rows are addressed as X + row * rowStride (768 for all nodes,
// 256*768 to gather per-graph root rows). 8 waves per block; each wave owns a
// 16-row M tile and all 8 N tiles of HID=128; K=768 in 24 steps of 32.
// B fragments are double-buffered across the unrolled j loop; raw A floats are
// double-buffered across the t loop, so WMMAs never wait on fresh loads.
// ---------------------------------------------------------------------------
template <bool DUAL, bool RELU>
__global__ void __launch_bounds__(256)
gemm_kernel(const float* __restrict__ X, long long rowStride,
            const _Float16* __restrict__ Bl, const _Float16* __restrict__ Br,
            float* __restrict__ Yl, float* __restrict__ Yr,
            const float* __restrict__ bias) {
  const int lane  = threadIdx.x & 31;
  const int wave  = threadIdx.x >> 5;   // 0..7
  const int m16   = lane & 15;          // A-fragment row within M tile / D column
  const int khalf = lane >> 4;          // 0 or 1

  const long long row0 = (long long)blockIdx.x * 128 + wave * 16;
  // 16-bit A 16x32 layout: lane (m16,khalf) needs contiguous f32 chunks
  // [t*32 + khalf*8 .. +7] and [t*32 + 16 + khalf*8 .. +7] of its row.
  const float* xrow = X + (row0 + m16) * rowStride + khalf * 8;

  v8f accL[NTILES];
  v8f accR[NTILES];
#pragma unroll
  for (int j = 0; j < NTILES; ++j) { accL[j] = (v8f){}; accR[j] = (v8f){}; }

  const v8h* BlV = (const v8h*)Bl;
  const v8h* BrV = (const v8h*)Br;

  // --- prologue: B fragment 0 and raw A floats for t=0 ---
  v8h bufA[4], bufB[4];
  {
    size_t bo = (size_t)lane * 2;
    bufA[0] = BlV[bo];
    bufA[1] = BlV[bo + 1];
    if constexpr (DUAL) { bufA[2] = BrV[bo]; bufA[3] = BrV[bo + 1]; }
  }
  float4 fa0 = *(const float4*)(xrow);
  float4 fa1 = *(const float4*)(xrow + 4);
  float4 fa2 = *(const float4*)(xrow + 16);
  float4 fa3 = *(const float4*)(xrow + 20);

  for (int t = 0; t < KSTEPS; ++t) {
    // Issue next-step A loads first so they overlap the 8-16 WMMAs below.
    const int tn = (t + 1 < KSTEPS) ? t + 1 : 0;
    const float* pn = xrow + tn * 32;
    float4 fn0 = *(const float4*)(pn);
    float4 fn1 = *(const float4*)(pn + 4);
    float4 fn2 = *(const float4*)(pn + 16);
    float4 fn3 = *(const float4*)(pn + 20);

    union { v16h v; _Float16 e[16]; } A;
    A.e[0]  = (_Float16)fa0.x;  A.e[1]  = (_Float16)fa0.y;
    A.e[2]  = (_Float16)fa0.z;  A.e[3]  = (_Float16)fa0.w;
    A.e[4]  = (_Float16)fa1.x;  A.e[5]  = (_Float16)fa1.y;
    A.e[6]  = (_Float16)fa1.z;  A.e[7]  = (_Float16)fa1.w;
    A.e[8]  = (_Float16)fa2.x;  A.e[9]  = (_Float16)fa2.y;
    A.e[10] = (_Float16)fa2.z;  A.e[11] = (_Float16)fa2.w;
    A.e[12] = (_Float16)fa3.x;  A.e[13] = (_Float16)fa3.y;
    A.e[14] = (_Float16)fa3.z;  A.e[15] = (_Float16)fa3.w;

#pragma unroll
    for (int j = 0; j < NTILES; ++j) {
      // Double-buffer: consume cb, load fragment (t*8+j+1) into nb.
      v8h* cb = (j & 1) ? bufB : bufA;
      v8h* nb = (j & 1) ? bufA : bufB;
      int nxt = t * NTILES + j + 1;
      if (nxt >= NFRAGS) nxt = 0;             // harmless dummy reload at the end
      size_t bo = ((size_t)nxt * 32 + lane) * 2;
      nb[0] = BlV[bo];
      nb[1] = BlV[bo + 1];
      if constexpr (DUAL) { nb[2] = BrV[bo]; nb[3] = BrV[bo + 1]; }

      union { v16h v; v8h h[2]; } B;
      B.h[0] = cb[0];
      B.h[1] = cb[1];
      accL[j] = __builtin_amdgcn_wmma_f32_16x16x32_f16(
          false, A.v, false, B.v, (short)0, accL[j], false, false);
      if constexpr (DUAL) {
        B.h[0] = cb[2];
        B.h[1] = cb[3];
        accR[j] = __builtin_amdgcn_wmma_f32_16x16x32_f16(
            false, A.v, false, B.v, (short)0, accR[j], false, false);
      }
    }
    fa0 = fn0; fa1 = fn1; fa2 = fn2; fa3 = fn3;
  }

  // ---- Epilogue. D layout: lane l -> column n = l%16, VGPR r -> row r + 8*(l/16).
#pragma unroll
  for (int j = 0; j < NTILES; ++j) {
    int col = j * 16 + m16;
    float b = RELU ? bias[col] : 0.0f;
#pragma unroll
    for (int r = 0; r < 8; ++r) {
      long long orow = row0 + r + 8 * khalf;
      float vL = accL[j][r];
      if constexpr (RELU) { vL += b; vL = vL > 0.0f ? vL : 0.0f; }
      Yl[orow * HID + col] = vL;
      if constexpr (DUAL) Yr[orow * HID + col] = accR[j][r];
    }
  }
}

// ---------------------------------------------------------------------------
// Edge scatter on projected features: msg[dst] += y_l[src]; cnt[dst] += 1.
// One wave per edge, 4 floats per lane.
// ---------------------------------------------------------------------------
__global__ void __launch_bounds__(256)
scatter_kernel(const int* __restrict__ src, const int* __restrict__ dst,
               const float* __restrict__ yl, float* __restrict__ msg,
               float* __restrict__ cnt, int nE) {
  int e = blockIdx.x * 8 + (threadIdx.x >> 5);
  if (e >= nE) return;
  int lane = threadIdx.x & 31;
  int s = src[e];
  int d = dst[e];
  float4 v = *(const float4*)(yl + (size_t)s * HID + lane * 4);
  float* mp = msg + (size_t)d * HID + lane * 4;
  atomicAdd(mp + 0, v.x);
  atomicAdd(mp + 1, v.y);
  atomicAdd(mp + 2, v.z);
  atomicAdd(mp + 3, v.w);
  if (lane == 0) atomicAdd(cnt + d, 1.0f);
}

// ---------------------------------------------------------------------------
// Per-graph: h = relu(msg/max(cnt,1) + b_l + y_r), pooled = max over 256 nodes.
// One block of 128 threads (one per feature) per graph; batch is contiguous.
// ---------------------------------------------------------------------------
__global__ void __launch_bounds__(128)
pool_kernel(const float* __restrict__ msg, const float* __restrict__ cnt,
            const float* __restrict__ yr, const float* __restrict__ b_l,
            float* __restrict__ pooled) {
  int g = blockIdx.x;
  int f = threadIdx.x;
  float bm = b_l[f];
  float best = 0.0f;  // relu(h) >= 0 and every graph has 256 nodes
  for (int i = 0; i < NODES_PER_GRAPH; ++i) {
    size_t node = (size_t)g * NODES_PER_GRAPH + i;
    float c = cnt[node];
    c = c > 1.0f ? c : 1.0f;
    float h = msg[node * HID + f] / c + bm + yr[node * HID + f];
    h = h > 0.0f ? h : 0.0f;
    best = h > best ? h : best;
  }
  pooled[(size_t)g * HID + f] = best;
}

// ---------------------------------------------------------------------------
// Head: z = relu(concat(pooled, news) @ W1 + b1); out = log_softmax(z @ W2 + b2).
// One block of 128 threads per graph.
// ---------------------------------------------------------------------------
__global__ void __launch_bounds__(128)
head_kernel(const float* __restrict__ pooled, const float* __restrict__ news,
            const float* __restrict__ W1, const float* __restrict__ b1,
            const float* __restrict__ W2, const float* __restrict__ b2,
            float* __restrict__ out) {
  __shared__ float in[2 * HID];
  __shared__ float z[HID];
  __shared__ float logits[2];
  int g = blockIdx.x;
  int t = threadIdx.x;
  in[t]       = pooled[(size_t)g * HID + t];
  in[t + HID] = news[(size_t)g * HID + t];
  __syncthreads();
  float acc = b1[t];
  for (int k = 0; k < 2 * HID; ++k) acc += in[k] * W1[k * HID + t];
  z[t] = acc > 0.0f ? acc : 0.0f;
  __syncthreads();
  if (t < 2) {
    float a = b2[t];
    for (int k = 0; k < HID; ++k) a += z[k] * W2[k * 2 + t];
    logits[t] = a;
  }
  __syncthreads();
  if (t == 0) {
    float l0 = logits[0], l1 = logits[1];
    float m = l0 > l1 ? l0 : l1;
    float lse = m + logf(expf(l0 - m) + expf(l1 - m));
    out[(size_t)g * 2 + 0] = l0 - lse;
    out[(size_t)g * 2 + 1] = l1 - lse;
  }
}

// ---------------------------------------------------------------------------
extern "C" void kernel_launch(void* const* d_in, const int* in_sizes, int n_in,
                              void* d_out, int out_size, void* d_ws, size_t ws_size,
                              hipStream_t stream) {
  const float* x    = (const float*)d_in[0];
  const int*   ei   = (const int*)d_in[1];     // [2, N_EDGES] flat
  // d_in[2] = batch (implicit: contiguous, 256 nodes per graph)
  const float* W_l  = (const float*)d_in[3];
  const float* b_l  = (const float*)d_in[4];
  const float* W_r  = (const float*)d_in[5];
  const float* W0   = (const float*)d_in[6];
  const float* b0   = (const float*)d_in[7];
  const float* W1   = (const float*)d_in[8];
  const float* b1   = (const float*)d_in[9];
  const float* W2   = (const float*)d_in[10];
  const float* b2   = (const float*)d_in[11];
  float* out = (float*)d_out;

  const int* src = ei;
  const int* dst = ei + N_EDGES;

  // Workspace layout (all sizes already 256B multiples).
  char* ws = (char*)d_ws;
  size_t off = 0;
  float*    yl     = (float*)(ws + off); off += (size_t)N_NODES * HID * 4;   // 256 MB
  float*    yr     = (float*)(ws + off); off += (size_t)N_NODES * HID * 4;   // 256 MB
  float*    msg    = (float*)(ws + off); off += (size_t)N_NODES * HID * 4;   // 256 MB
  float*    cnt    = (float*)(ws + off); off += (size_t)N_NODES * 4;         // 2 MB
  float*    pooled = (float*)(ws + off); off += (size_t)NUM_GRAPHS * HID * 4;
  float*    news   = (float*)(ws + off); off += (size_t)NUM_GRAPHS * HID * 4;
  _Float16* swzL   = (_Float16*)(ws + off); off += (size_t)F_IN * HID * 2;
  _Float16* swzR   = (_Float16*)(ws + off); off += (size_t)F_IN * HID * 2;
  _Float16* swz0   = (_Float16*)(ws + off); off += (size_t)F_IN * HID * 2;

  // 1) Pre-swizzle weights to f16 B-fragment layout.
  int swzElems = NFRAGS * 32 * 16;
  int swzBlocks = (swzElems + 255) / 256;
  swizzle_w_kernel<<<swzBlocks, 256, 0, stream>>>(W_l, swzL);
  swizzle_w_kernel<<<swzBlocks, 256, 0, stream>>>(W_r, swzR);
  swizzle_w_kernel<<<swzBlocks, 256, 0, stream>>>(W0, swz0);

  // 2) Zero msg + cnt (contiguous).
  long long nZero = (long long)N_NODES * HID + N_NODES;
  zero_f32_kernel<<<4096, 256, 0, stream>>>(msg, nZero);

  // 3) y_l = x @ W_l, y_r = x @ W_r (single pass over x).
  gemm_kernel<true, false><<<N_NODES / 128, 256, 0, stream>>>(
      x, (long long)F_IN, swzL, swzR, yl, yr, nullptr);

  // 4) news = relu(x[roots] @ W0 + b0); root of graph g is node g*256.
  gemm_kernel<false, true><<<NUM_GRAPHS / 128, 256, 0, stream>>>(
      x, (long long)NODES_PER_GRAPH * F_IN, swz0, nullptr, news, nullptr, b0);

  // 5) Aggregate projected messages along edges.
  scatter_kernel<<<N_EDGES / 8, 256, 0, stream>>>(src, dst, yl, msg, cnt, N_EDGES);

  // 6) h = relu(mean + b_l + y_r), per-graph max pool.
  pool_kernel<<<NUM_GRAPHS, 128, 0, stream>>>(msg, cnt, yr, b_l, pooled);

  // 7) MLP head + log_softmax.
  head_kernel<<<NUM_GRAPHS, 128, 0, stream>>>(pooled, news, W1, b1, W2, b2, out);

  (void)in_sizes; (void)n_in; (void)out_size; (void)ws_size;
}
